// DeepMOI_29996051595782
// MI455X (gfx1250) — compile-verified, matrix operand order checked by reading, not run on hardware
//
#include <hip/hip_runtime.h>
#include <hip/hip_bf16.h>
#include <math.h>

typedef __attribute__((ext_vector_type(16))) _Float16 v16h;
typedef __attribute__((ext_vector_type(8)))  _Float16 v8h;
typedef __attribute__((ext_vector_type(8)))  float    v8f;

#define N_NODES 20000
#define N_EDGES 640000
#define P_PATH  512
#define S_SUB   64
#define D_DIM   256

// ---------------------------------------------------------------------------
// agg[n, 0:din] = h[n], agg[n, din:Kp] = 0   (folds the "h + sum" add)
__global__ void k_init_agg(float* __restrict__ agg, const float* __restrict__ h,
                           int din, int Kp, long total) {
    long i = (long)blockIdx.x * blockDim.x + threadIdx.x;
    long stride = (long)gridDim.x * blockDim.x;
    for (; i < total; i += stride) {
        long n = i / Kp; int k = (int)(i - n * Kp);
        agg[i] = (k < din) ? h[n * din + k] : 0.0f;
    }
}

// agg[dst] += h[src]; one thread per (edge, feature) so consecutive lanes
// touch consecutive features -> coalesced b32 loads + global_atomic_add_f32
__global__ void k_scatter(float* __restrict__ agg, const float* __restrict__ h,
                          const int* __restrict__ src, const int* __restrict__ dst,
                          int din, int Kp, long total) {
    long i = (long)blockIdx.x * blockDim.x + threadIdx.x;
    long stride = (long)gridDim.x * blockDim.x;
    for (; i < total; i += stride) {
        long e = i / din; int k = (int)(i - e * din);
        atomicAdd(&agg[(long)dst[e] * Kp + k], h[(long)src[e] * din + k]);
    }
}

__global__ void k_f2h(_Float16* __restrict__ o, const float* __restrict__ x, long total) {
    long i = (long)blockIdx.x * blockDim.x + threadIdx.x;
    long stride = (long)gridDim.x * blockDim.x;
    for (; i < total; i += stride) o[i] = (_Float16)x[i];
}

// Wh [Npad64, Kp] f16 from W [Nout, Kin] f32, zero-padded (rows padded to x64
// so the 16x64-tile GEMM always reads defined zeros)
__global__ void k_cvtw(_Float16* __restrict__ Wh, const float* __restrict__ W,
                       int Nout, int Kin, int Kp, long total) {
    long i = (long)blockIdx.x * blockDim.x + threadIdx.x;
    long stride = (long)gridDim.x * blockDim.x;
    for (; i < total; i += stride) {
        long j = i / Kp; int k = (int)(i - j * Kp);
        Wh[i] = (j < Nout && k < Kin) ? (_Float16)W[j * Kin + k] : (_Float16)0.0f;
    }
}

// ---------------------------------------------------------------------------
// C[M,N] = act(A[M,Kp] * B[N,Kp]^T + bias[N]).  f16 inputs, f32 accumulate.
// One wave per 16x64 output tile: 1 A fragment + 4 B fragments per K-step,
// 4 independent back-to-back v_wmma (no D->A/B hazard NOPs).
// grid = (Npad64/64, M/16), block = 32.
// A-fragment per ISA 7.12.2: lane L (m = L&15, half = L>>4) holds
//   K = k0 + half*8 + {0..7}  and  K = k0 + 16 + half*8 + {0..7}
// (two contiguous 16-byte groups). B assumed symmetric with n = L&15.
__global__ void k_wmma_gemm(const _Float16* __restrict__ A,
                            const _Float16* __restrict__ B,
                            const float* __restrict__ bias,
                            float* __restrict__ C,
                            int Kp, int Nvalid, int Ns, int act) {
    const int lane = threadIdx.x;
    const int half = lane >> 4;
    const int l16  = lane & 15;
    const long mBase = (long)blockIdx.y * 16;
    const long nBase = (long)blockIdx.x * 64;

    const _Float16* arow = A + (mBase + l16) * (long)Kp + half * 8;
    const _Float16* brow = B + (nBase + l16) * (long)Kp + half * 8;
    const long bstep = 16L * Kp;                 // next 16-col B tile

    v8f acc0 = {}, acc1 = {}, acc2 = {}, acc3 = {};
    for (int k0 = 0; k0 < Kp; k0 += 32) {
        if (k0 + 32 < Kp) {                      // gfx1250 global_prefetch_b8
            __builtin_prefetch(arow + 32, 0, 3);
            __builtin_prefetch(brow + 32, 0, 3);
        }
        v8h alo = *(const v8h*)(arow);
        v8h ahi = *(const v8h*)(arow + 16);
        v16h av;
#pragma unroll
        for (int i = 0; i < 8; ++i) { av[i] = alo[i]; av[i + 8] = ahi[i]; }

        v16h bv0, bv1, bv2, bv3;
#pragma unroll
        for (int t = 0; t < 4; ++t) {
            const _Float16* bp = brow + t * bstep;
            v8h blo = *(const v8h*)(bp);
            v8h bhi = *(const v8h*)(bp + 16);
            v16h bv;
#pragma unroll
            for (int i = 0; i < 8; ++i) { bv[i] = blo[i]; bv[i + 8] = bhi[i]; }
            if (t == 0) bv0 = bv; else if (t == 1) bv1 = bv;
            else if (t == 2) bv2 = bv; else bv3 = bv;
        }
        acc0 = __builtin_amdgcn_wmma_f32_16x16x32_f16(false, av, false, bv0, (short)0, acc0, false, false);
        acc1 = __builtin_amdgcn_wmma_f32_16x16x32_f16(false, av, false, bv1, (short)0, acc1, false, false);
        acc2 = __builtin_amdgcn_wmma_f32_16x16x32_f16(false, av, false, bv2, (short)0, acc2, false, false);
        acc3 = __builtin_amdgcn_wmma_f32_16x16x32_f16(false, av, false, bv3, (short)0, acc3, false, false);
        arow += 32;
        brow += 32;
    }

#pragma unroll
    for (int t = 0; t < 4; ++t) {
        v8f acc = (t == 0) ? acc0 : (t == 1) ? acc1 : (t == 2) ? acc2 : acc3;
        const long col = nBase + t * 16 + l16;
        if (col < Nvalid) {
            float bv0 = bias ? bias[col] : 0.0f;
#pragma unroll
            for (int r = 0; r < 8; ++r) {
                long row = mBase + r + half * 8; // C layout: VGPR r -> M = r (+8 hi half)
                float v = acc[r] + bv0;
                if (act) v = v > 0.0f ? v : 0.0f;
                C[row * (long)Ns + col] = v;
            }
        }
    }
}

// ---------------------------------------------------------------------------
// Set2Set helpers
__global__ void k_build_wcat(_Float16* __restrict__ Wcat,
                             const float* __restrict__ Wih,
                             const float* __restrict__ Whh) {
    long i = (long)blockIdx.x * blockDim.x + threadIdx.x;
    long stride = (long)gridDim.x * blockDim.x;
    const long per = 1024L * 768;
    for (; i < 3 * per; i += stride) {
        long k = i / per, rem = i - k * per;
        long j = rem / 768; int c = (int)(rem - j * 768);
        float v = (c < 512) ? Wih[(k * 1024 + j) * 512 + c]
                            : Whh[(k * 1024 + j) * 256 + (c - 512)];
        Wcat[i] = (_Float16)v;
    }
}

__global__ void k_build_bcat(float* __restrict__ bcat,
                             const float* __restrict__ bih,
                             const float* __restrict__ bhh) {
    int i = blockIdx.x * blockDim.x + threadIdx.x;
    if (i < 3 * 1024) bcat[i] = bih[i] + bhh[i];
}

__global__ void k_build_xcat(_Float16* __restrict__ X,
                             const float* __restrict__ qstar,
                             const float* __restrict__ hs) {
    int i = blockIdx.x * blockDim.x + threadIdx.x;
    if (i >= P_PATH * 768) return;
    int p = i / 768, c = i - p * 768;
    float v = (c < 512) ? qstar[p * 512 + c] : hs[p * 256 + (c - 512)];
    X[i] = (_Float16)v;
}

__device__ __forceinline__ float sigf(float x) { return 1.0f / (1.0f + expf(-x)); }

// gates [P,4D] layout [i,f,g,o]; update cs, hs in place
__global__ void k_lstm(const float* __restrict__ gates,
                       float* __restrict__ cs, float* __restrict__ hs) {
    int i = blockIdx.x * blockDim.x + threadIdx.x;
    if (i >= P_PATH * D_DIM) return;
    int p = i / D_DIM, d = i - p * D_DIM;
    const float* g = gates + (long)p * 4 * D_DIM;
    float ig = g[d], fg = g[D_DIM + d], gg = g[2 * D_DIM + d], og = g[3 * D_DIM + d];
    float c = sigf(fg) * cs[i] + sigf(ig) * tanhf(gg);
    cs[i] = c;
    hs[i] = sigf(og) * tanhf(c);
}

// per-pathway attention: scores = x·q, softmax over S, r = Σ alpha x
// qstar[p] = [q | r]; optionally also written into readout slot kmod.
__global__ void k_attention(const float* __restrict__ h4,
                            const int* __restrict__ path_nodes,
                            const float* __restrict__ hs,
                            float* __restrict__ qstar,
                            float* __restrict__ readout,
                            int kmod, int writeReadout) {
    __shared__ float sc[S_SUB];
    __shared__ float red[2];
    const int p = blockIdx.x;
    const int t = threadIdx.x;                  // 256 threads
    const int* pn = path_nodes + p * S_SUB;
    const float* q = hs + (long)p * D_DIM;

    if (t < S_SUB) {
        const float* xr = h4 + (long)pn[t] * D_DIM;
        float a = 0.0f;
        for (int d = 0; d < D_DIM; ++d) a += xr[d] * q[d];
        sc[t] = a;
    }
    __syncthreads();
    if (t == 0) {
        float mx = sc[0];
        for (int s = 1; s < S_SUB; ++s) mx = fmaxf(mx, sc[s]);
        float sm = 0.0f;
        for (int s = 0; s < S_SUB; ++s) sm += expf(sc[s] - mx);
        red[0] = mx; red[1] = sm;
    }
    __syncthreads();
    if (t < S_SUB) sc[t] = expf(sc[t] - red[0]) / red[1];
    __syncthreads();

    float r = 0.0f;                              // each thread owns d = t (D = 256)
    for (int s = 0; s < S_SUB; ++s)
        r += sc[s] * h4[(long)pn[s] * D_DIM + t];

    float qv = q[t];
    qstar[(long)p * 512 + t]        = qv;
    qstar[(long)p * 512 + 256 + t]  = r;
    if (writeReadout) {
        float* ro = readout + (long)p * 1536 + kmod * 512;
        ro[t]       = qv;
        ro[256 + t] = r;
    }
}

// ---------------------------------------------------------------------------
// Final MLP head, single block of 512 threads
__global__ void k_head(const float* __restrict__ readout,
                       const float* __restrict__ Wg, const float* __restrict__ bg,
                       const float* __restrict__ Wl1, const float* __restrict__ bl1,
                       const float* __restrict__ Wl2, const float* __restrict__ bl2,
                       const float* __restrict__ Wl3, const float* __restrict__ bl3,
                       float* __restrict__ out) {
    __shared__ float xg[P_PATH];
    __shared__ float z1[256];
    __shared__ float z2[64];
    const int t = threadIdx.x;
    {   // xg[p] = readout[p]·Wg + bg
        float a = bg[0];
        const float* rr = readout + (long)t * 1536;
        for (int c = 0; c < 1536; ++c) a += rr[c] * Wg[c];
        xg[t] = a;
    }
    __syncthreads();
    if (t < 256) {
        float a = bl1[t];
        const float* w = Wl1 + (long)t * 512;
        for (int p = 0; p < P_PATH; ++p) a += xg[p] * w[p];
        z1[t] = tanhf(a);
    }
    __syncthreads();
    if (t < 64) {
        float a = bl2[t];
        const float* w = Wl2 + (long)t * 256;
        for (int j = 0; j < 256; ++j) a += z1[j] * w[j];
        z2[t] = fmaxf(a, 0.0f);
    }
    __syncthreads();
    if (t == 0) {
        float a = bl3[0];
        for (int j = 0; j < 64; ++j) a += z2[j] * Wl3[j];
        out[0] = 1.0f / (1.0f + expf(-a));
    }
}

// ---------------------------------------------------------------------------
extern "C" void kernel_launch(void* const* d_in, const int* in_sizes, int n_in,
                              void* d_out, int out_size, void* d_ws, size_t ws_size,
                              hipStream_t stream) {
    const float* h0   = (const float*)d_in[0];
    const int*   src  = (const int*)d_in[1];
    const int*   dst  = (const int*)d_in[2];
    const int*   path = (const int*)d_in[3];
    const float* W[4]  = {(const float*)d_in[4], (const float*)d_in[6],
                          (const float*)d_in[8], (const float*)d_in[10]};
    const float* Bl[4] = {(const float*)d_in[5], (const float*)d_in[7],
                          (const float*)d_in[9], (const float*)d_in[11]};
    const float* Wih = (const float*)d_in[12];
    const float* Whh = (const float*)d_in[13];
    const float* bih = (const float*)d_in[14];
    const float* bhh = (const float*)d_in[15];
    const float* Wg  = (const float*)d_in[16];
    const float* bg  = (const float*)d_in[17];
    const float* Wl1 = (const float*)d_in[18]; const float* bl1 = (const float*)d_in[19];
    const float* Wl2 = (const float*)d_in[20]; const float* bl2 = (const float*)d_in[21];
    const float* Wl3 = (const float*)d_in[22]; const float* bl3 = (const float*)d_in[23];

    // workspace carve-out (256B aligned)
    char* base = (char*)d_ws;
    size_t off = 0;
    auto take = [&](size_t b) -> char* {
        char* p = base + off;
        off = (off + b + 255) & ~(size_t)255;
        return p;
    };
    float*     agg   = (float*)    take((size_t)N_NODES * 128 * 4);
    _Float16*  xh    = (_Float16*) take((size_t)N_NODES * 128 * 2);
    _Float16*  Wh    = (_Float16*) take((size_t)256 * 128 * 2);
    float*     hA    = (float*)    take((size_t)N_NODES * 256 * 4);
    float*     hB    = (float*)    take((size_t)N_NODES * 256 * 4);
    _Float16*  Wcat  = (_Float16*) take((size_t)3 * 1024 * 768 * 2);
    float*     bcat  = (float*)    take((size_t)3 * 1024 * 4);
    float*     qstar = (float*)    take((size_t)P_PATH * 512 * 4);
    float*     hsb   = (float*)    take((size_t)P_PATH * 256 * 4);
    float*     csb   = (float*)    take((size_t)P_PATH * 256 * 4);
    float*     gates = (float*)    take((size_t)P_PATH * 1024 * 4);
    _Float16*  Xcat  = (_Float16*) take((size_t)P_PATH * 768 * 2);
    float*     rdout = (float*)    take((size_t)P_PATH * 1536 * 4);

    // ---- 4x GINConv ----
    const int dims[5] = {8, 24, 64, 128, 256};
    const float* hin = h0;
    float* outs[4] = {hA, hB, hA, hB};
    for (int l = 0; l < 4; ++l) {
        int din = dims[l], dout = dims[l + 1];
        int Kp     = (din  + 31) & ~31;
        int Npad64 = (dout + 63) & ~63;
        long tot = (long)N_NODES * Kp;
        k_init_agg<<<(int)((tot + 255) / 256), 256, 0, stream>>>(agg, hin, din, Kp, tot);
        long etot = (long)N_EDGES * din;
        k_scatter<<<(int)((etot + 255) / 256), 256, 0, stream>>>(agg, hin, src, dst, din, Kp, etot);
        k_f2h<<<(int)((tot + 255) / 256), 256, 0, stream>>>(xh, agg, tot);
        long wt = (long)Npad64 * Kp;
        k_cvtw<<<(int)((wt + 255) / 256), 256, 0, stream>>>(Wh, W[l], dout, din, Kp, wt);
        dim3 g(Npad64 / 64, N_NODES / 16);
        k_wmma_gemm<<<g, 32, 0, stream>>>(xh, Wh, Bl[l], outs[l], Kp, dout, dout, 1);
        hin = outs[l];
    }
    const float* h4 = outs[3];   // [N_NODES, 256]

    // ---- 3x Set2Set (2 iterations each) ----
    k_build_wcat<<<(int)((3L * 1024 * 768 + 255) / 256), 256, 0, stream>>>(Wcat, Wih, Whh);
    k_build_bcat<<<(3 * 1024 + 255) / 256, 256, 0, stream>>>(bcat, bih, bhh);
    for (int k = 0; k < 3; ++k) {
        hipMemsetAsync(qstar, 0, (size_t)P_PATH * 512 * 4, stream);
        hipMemsetAsync(hsb,   0, (size_t)P_PATH * 256 * 4, stream);
        hipMemsetAsync(csb,   0, (size_t)P_PATH * 256 * 4, stream);
        for (int t = 0; t < 2; ++t) {
            k_build_xcat<<<(P_PATH * 768 + 255) / 256, 256, 0, stream>>>(Xcat, qstar, hsb);
            dim3 g(1024 / 64, P_PATH / 16);
            k_wmma_gemm<<<g, 32, 0, stream>>>(Xcat, Wcat + (size_t)k * 1024 * 768,
                                              bcat + k * 1024, gates, 768, 1024, 1024, 0);
            k_lstm<<<(P_PATH * 256 + 255) / 256, 256, 0, stream>>>(gates, csb, hsb);
            k_attention<<<P_PATH, 256, 0, stream>>>(h4, path, hsb, qstar, rdout, k, t == 1);
        }
    }

    // ---- head ----
    k_head<<<1, 512, 0, stream>>>(rdout, Wg, bg, Wl1, bl1, Wl2, bl2, Wl3, bl3, (float*)d_out);
}
